// SwinTransformerBlock_63050119905818
// MI455X (gfx1250) — matrix operand order repaired and checked by reading, not compiled, not run on hardware
//
#include <hip/hip_runtime.h>
#include <hip/hip_bf16.h>
#include <math.h>

// ---------------- constants ----------------
#define BB   16
#define HH_  56
#define WW_  56
#define CC   256
#define WSZ  7
#define SSH  3
#define NHD  8
#define HDD  32
#define LW   49          // WS*WS
#define HID_ 1024
#define NWIN 64          // (H/WS)*(W/WS)
#define TOTW (BB*NWIN)   // 1024 windows
#define MTOT (TOTW*LW)   // 50176 tokens (divisible by 16)

typedef __attribute__((ext_vector_type(16))) _Float16 v16h;
typedef __attribute__((ext_vector_type(8)))  float    v8f;

// ---------------- WMMA fragment loaders (gfx1250 wave32 layouts) ----------------
// A: 16x32 f16, row-major source (lda elems). lane m=lane&15, kh=lane>>4.
__device__ __forceinline__ v16h load_frag_a(const _Float16* A, int lda, int lane) {
    int m = lane & 15, kh = lane >> 4;
    union { uint4 u[2]; v16h h; } f;
    const _Float16* p = A + (size_t)m * lda;
    f.u[0] = *(const uint4*)(p + kh * 8);
    f.u[1] = *(const uint4*)(p + 16 + kh * 8);
    return f.h;
}
// B: 32x16 f16 from [N][K] row-major source (ldb elems). lane n=lane&15 holds
// K [kh*8, kh*8+8) then [16+kh*8, 16+kh*8+8), kh=lane>>4.
__device__ __forceinline__ v16h load_frag_b(const _Float16* Bw, int ldb, int lane) {
    int n = lane & 15, kh = lane >> 4;
    union { uint4 u[2]; v16h h; } f;
    const _Float16* p = Bw + (size_t)n * ldb;
    f.u[0] = *(const uint4*)(p + kh * 8);
    f.u[1] = *(const uint4*)(p + 16 + kh * 8);
    return f.h;
}
// one wave: 16(M) x 64(N) tile, K loop in steps of 32
__device__ __forceinline__ void gemm_tile_16x64(const _Float16* A, int lda,
                                                const _Float16* Bw, int ldb,
                                                int K, int lane, v8f acc[4]) {
    for (int k0 = 0; k0 < K; k0 += 32) {
        v16h a = load_frag_a(A + k0, lda, lane);
#pragma unroll
        for (int j = 0; j < 4; ++j) {
            v16h b = load_frag_b(Bw + (size_t)(j * 16) * ldb + k0, ldb, lane);
            acc[j] = __builtin_amdgcn_wmma_f32_16x16x32_f16(
                false, a, false, b, (short)0, acc[j], false, false);
        }
    }
}

// ---------------- kernel 0: f32 -> f16 weight convert ----------------
__global__ void k_cvt(const float* in, _Float16* out, int n) {
    int i = blockIdx.x * 256 + threadIdx.x;
    if (i < n) out[i] = (_Float16)in[i];
}

// ---------------- kernel 1: depthwise 3x3 APE conv + bias + residual ----------------
__global__ void k_conv(const float* __restrict__ x, const float* __restrict__ apew,
                       const float* __restrict__ apeb, float* __restrict__ xs) {
    size_t gid = (size_t)blockIdx.x * 256 + threadIdx.x;
    int c = (int)(gid & 255);
    size_t pix = gid >> 8;
    int w0 = (int)(pix % WW_);
    size_t t2 = pix / WW_;
    int h0 = (int)(t2 % HH_);
    int b  = (int)(t2 / HH_);
    float sum = x[gid];
#pragma unroll
    for (int dh = -1; dh <= 1; ++dh)
#pragma unroll
        for (int dw = -1; dw <= 1; ++dw) {
            int hh = h0 + dh, ww = w0 + dw;
            if (hh >= 0 && hh < HH_ && ww >= 0 && ww < WW_)
                sum += x[(((size_t)b * HH_ + hh) * WW_ + ww) * CC + c] *
                       apew[((dh + 1) * 3 + (dw + 1)) * CC + c];
        }
    xs[gid] = sum + apeb[c];
}

// ---------------- kernel 2: LN1 + roll(-SS) + window partition -> f16 ----------------
__global__ void k_ln1win(const float* __restrict__ xs, const float* __restrict__ w,
                         const float* __restrict__ bp, _Float16* __restrict__ out) {
    __shared__ float red[256];
    int t = blockIdx.x, c = threadIdx.x;
    int win = t / LW, l = t - win * LW;
    int b = win >> 6, wi = win & 63;
    int wh = wi >> 3, ww = wi & 7;
    int i = l / WSZ, j = l - i * WSZ;
    int h0 = (wh * WSZ + i + SSH) % HH_;
    int w0 = (ww * WSZ + j + SSH) % WW_;
    const float* row = xs + (((size_t)b * HH_ + h0) * WW_ + w0) * CC;
    float v = row[c];
    red[c] = v; __syncthreads();
    for (int s = 128; s > 0; s >>= 1) { if (c < s) red[c] += red[c + s]; __syncthreads(); }
    float mean = red[0] / (float)CC; __syncthreads();
    float d = v - mean;
    red[c] = d * d; __syncthreads();
    for (int s = 128; s > 0; s >>= 1) { if (c < s) red[c] += red[c + s]; __syncthreads(); }
    float rstd = rsqrtf(red[0] / (float)CC + 1e-5f);
    out[(size_t)t * CC + c] = (_Float16)(d * rstd * w[c] + bp[c]);
}

// ---------------- kernel 3: QKV GEMM (M=50176,N=768,K=256) + bias, head scatter ----------------
__global__ void k_gemm_qkv(const _Float16* __restrict__ A, const _Float16* __restrict__ W,
                           const float* __restrict__ bias, _Float16* __restrict__ qkv) {
    int wid = blockIdx.x * 4 + (threadIdx.x >> 5);
    int lane = threadIdx.x & 31;
    int mt = wid / 12, nq = wid - mt * 12;
    v8f acc[4] = {};
    gemm_tile_16x64(A + (size_t)mt * 16 * CC, CC, W + (size_t)nq * 64 * CC, CC, CC, lane, acc);
    int nl = lane & 15, mh = lane >> 4;
#pragma unroll
    for (int j = 0; j < 4; ++j) {
        int n = nq * 64 + j * 16 + nl;
        int p = n >> 8, hh = (n >> 5) & 7, d = n & 31;
        float bv = bias[n];
#pragma unroll
        for (int v = 0; v < 8; ++v) {
            int m = mt * 16 + v + mh * 8;
            int win = m / LW, l = m - win * LW;
            size_t off = (((size_t)(p * TOTW + win) * NHD + hh) * LW + l) * HDD + d;
            qkv[off] = (_Float16)(acc[j][v] + bv);
        }
    }
}

// ---------------- kernel 4: continuous rel-pos bias table (169 x NH) ----------------
__global__ void k_rpb(const float* __restrict__ w1, const float* __restrict__ b1,
                      const float* __restrict__ w2, const float* __restrict__ b2,
                      float* __restrict__ table) {
    int o = blockIdx.x * 256 + threadIdx.x;
    if (o >= 169 * NHD) return;
    int idx = o >> 3, head = o & 7;
    int a = idx / 13, bq = idx - a * 13;
    float da = (float)(a - 6), db = (float)(bq - 6);
    float fx = (da >= 0.f) ? log1pf(da) : -log1pf(-da);
    float fy = (db >= 0.f) ? log1pf(db) : -log1pf(-db);
    float acc = b2[head];
    for (int j = 0; j < 512; ++j) {
        float h = fmaxf(w1[j * 2] * fx + w1[j * 2 + 1] * fy + b1[j], 0.f);
        acc += h * w2[head * 512 + j];
    }
    table[idx * NHD + head] = acc;
}

// ---------------- kernel 5: windowed cosine attention (one block per win,head) ----------------
__global__ void k_attn(const _Float16* __restrict__ qkv, const float* __restrict__ scale,
                       const float* __restrict__ table, _Float16* __restrict__ attn_out) {
    int win = blockIdx.x >> 3;
    int head = blockIdx.x & 7;
    int tid = threadIdx.x;
    int lane = tid & 31, wv = tid >> 5;

    __shared__ __align__(16) _Float16 sQ[64 * 32];
    __shared__ __align__(16) _Float16 sK[64 * 32];
    __shared__ __align__(16) _Float16 sVT[32 * 64];
    __shared__ float sS[64 * 64];
    __shared__ __align__(16) _Float16 sP[64 * 64];

    const _Float16* qg = qkv + (((size_t)(0 * TOTW + win) * NHD + head) * LW) * HDD;
    const _Float16* kg = qkv + (((size_t)(1 * TOTW + win) * NHD + head) * LW) * HDD;
    const _Float16* vg = qkv + (((size_t)(2 * TOTW + win) * NHD + head) * LW) * HDD;

    if (tid < 64) {
        int r = tid;
        if (r < LW) {
            float s = 0.f;
            for (int d = 0; d < HDD; ++d) { float x = (float)qg[r * HDD + d]; s += x * x; }
            float inv = 1.f / fmaxf(sqrtf(s), 1e-12f);
            for (int d = 0; d < HDD; ++d) sQ[r * 32 + d] = (_Float16)((float)qg[r * HDD + d] * inv);
            for (int d = 0; d < HDD; ++d) sVT[d * 64 + r] = vg[r * HDD + d];
        } else {
            for (int d = 0; d < HDD; ++d) sQ[r * 32 + d] = (_Float16)0.f;
            for (int d = 0; d < HDD; ++d) sVT[d * 64 + r] = (_Float16)0.f;
        }
    } else {
        int r = tid - 64;
        if (r < LW) {
            float s = 0.f;
            for (int d = 0; d < HDD; ++d) { float x = (float)kg[r * HDD + d]; s += x * x; }
            float inv = 1.f / fmaxf(sqrtf(s), 1e-12f);
            for (int d = 0; d < HDD; ++d) sK[r * 32 + d] = (_Float16)((float)kg[r * HDD + d] * inv);
        } else {
            for (int d = 0; d < HDD; ++d) sK[r * 32 + d] = (_Float16)0.f;
        }
    }
    __syncthreads();

    float esc = expf(-scale[head]);
    // S = q_hat @ k_hat^T : 64x64, K=32 -> one WMMA per tile
    v8f sacc[4] = {};
    {
        v16h a = load_frag_a(sQ + wv * 16 * 32, 32, lane);
#pragma unroll
        for (int j = 0; j < 4; ++j) {
            v16h b = load_frag_b(sK + j * 16 * 32, 32, lane);
            sacc[j] = __builtin_amdgcn_wmma_f32_16x16x32_f16(
                false, a, false, b, (short)0, sacc[j], false, false);
        }
    }
    int wi = win & 63, wh = wi >> 3, ww = wi & 7;
    int nl = lane & 15, mh = lane >> 4;
#pragma unroll
    for (int j = 0; j < 4; ++j) {
#pragma unroll
        for (int v = 0; v < 8; ++v) {
            int m = wv * 16 + v + mh * 8;
            int n = j * 16 + nl;
            float val;
            if (n >= LW) val = -1e30f;            // pad columns excluded from softmax
            else if (m >= LW) val = 0.f;          // pad rows (discarded later)
            else {
                int i1 = m / WSZ, j1 = m - i1 * WSZ;
                int i2 = n / WSZ, j2 = n - i2 * WSZ;
                int ri = (i1 - i2 + 6) * 13 + (j1 - j2 + 6);
                float bias = table[ri * NHD + head];
                int hs1 = wh * WSZ + i1, ws1 = ww * WSZ + j1;
                int hs2 = wh * WSZ + i2, ws2 = ww * WSZ + j2;
                int r1 = ((hs1 < 49) ? 0 : ((hs1 < 53) ? 1 : 2)) * 3 +
                         ((ws1 < 49) ? 0 : ((ws1 < 53) ? 1 : 2));
                int r2 = ((hs2 < 49) ? 0 : ((hs2 < 53) ? 1 : 2)) * 3 +
                         ((ws2 < 49) ? 0 : ((ws2 < 53) ? 1 : 2));
                float msk = (r1 == r2) ? 0.f : -100.f;
                val = sacc[j][v] * esc + bias + msk;
            }
            sS[m * 64 + n] = val;
        }
    }
    __syncthreads();

    // softmax per row over n<49, write f16 P (pad cols zero)
    if (tid < 64) {
        int r = tid;
        float mx = -1e30f;
        for (int n = 0; n < LW; ++n) mx = fmaxf(mx, sS[r * 64 + n]);
        float sum = 0.f;
        for (int n = 0; n < LW; ++n) sum += expf(sS[r * 64 + n] - mx);
        float inv = 1.f / sum;
        for (int n = 0; n < 64; ++n)
            sP[r * 64 + n] = (n < LW) ? (_Float16)(expf(sS[r * 64 + n] - mx) * inv)
                                      : (_Float16)0.f;
    }
    __syncthreads();

    // O = P @ V : M=64, N=32, K=64 (vT in LDS so B frags are contiguous)
    v8f oacc[2] = {};
#pragma unroll
    for (int k0 = 0; k0 < 64; k0 += 32) {
        v16h a = load_frag_a(sP + wv * 16 * 64 + k0, 64, lane);
#pragma unroll
        for (int j = 0; j < 2; ++j) {
            v16h b = load_frag_b(sVT + (size_t)(j * 16) * 64 + k0, 64, lane);
            oacc[j] = __builtin_amdgcn_wmma_f32_16x16x32_f16(
                false, a, false, b, (short)0, oacc[j], false, false);
        }
    }
#pragma unroll
    for (int j = 0; j < 2; ++j) {
#pragma unroll
        for (int v = 0; v < 8; ++v) {
            int m = wv * 16 + v + mh * 8;
            if (m < LW) {
                size_t t = (size_t)win * LW + m;
                attn_out[t * CC + head * HDD + j * 16 + nl] = (_Float16)(oacc[j][v]);
            }
        }
    }
}

// ---------------- kernel 6: proj GEMM + window-reverse + roll(+SS) + residual ----------------
__global__ void k_gemm_proj(const _Float16* __restrict__ A, const _Float16* __restrict__ W,
                            const float* __restrict__ bias, const float* __restrict__ xs,
                            float* __restrict__ x2) {
    int wid = blockIdx.x * 4 + (threadIdx.x >> 5);
    int lane = threadIdx.x & 31;
    int mt = wid >> 2, nq = wid & 3;
    v8f acc[4] = {};
    gemm_tile_16x64(A + (size_t)mt * 16 * CC, CC, W + (size_t)nq * 64 * CC, CC, CC, lane, acc);
    int nl = lane & 15, mh = lane >> 4;
#pragma unroll
    for (int j = 0; j < 4; ++j) {
        int n = nq * 64 + j * 16 + nl;
        float bv = bias[n];
#pragma unroll
        for (int v = 0; v < 8; ++v) {
            int m = mt * 16 + v + mh * 8;
            int win = m / LW, l = m - win * LW;
            int b = win >> 6, wi = win & 63, wh = wi >> 3, ww = wi & 7;
            int i = l / WSZ, jj = l - i * WSZ;
            int h0 = (wh * WSZ + i + SSH) % HH_;
            int w0 = (ww * WSZ + jj + SSH) % WW_;
            size_t idx = (((size_t)b * HH_ + h0) * WW_ + w0) * CC + n;
            x2[idx] = xs[idx] + acc[j][v] + bv;
        }
    }
}

// ---------------- kernel 7: LN2 -> f16 (natural raster token order) ----------------
__global__ void k_ln2(const float* __restrict__ x2, const float* __restrict__ w,
                      const float* __restrict__ bp, _Float16* __restrict__ out) {
    __shared__ float red[256];
    int t = blockIdx.x, c = threadIdx.x;
    float v = x2[(size_t)t * CC + c];
    red[c] = v; __syncthreads();
    for (int s = 128; s > 0; s >>= 1) { if (c < s) red[c] += red[c + s]; __syncthreads(); }
    float mean = red[0] / (float)CC; __syncthreads();
    float d = v - mean;
    red[c] = d * d; __syncthreads();
    for (int s = 128; s > 0; s >>= 1) { if (c < s) red[c] += red[c + s]; __syncthreads(); }
    float rstd = rsqrtf(red[0] / (float)CC + 1e-5f);
    out[(size_t)t * CC + c] = (_Float16)(d * rstd * w[c] + bp[c]);
}

// ---------------- kernel 8: fc1 GEMM (N=1024,K=256) + exact GELU ----------------
__global__ void k_gemm_fc1(const _Float16* __restrict__ A, const _Float16* __restrict__ W,
                           const float* __restrict__ bias, _Float16* __restrict__ g) {
    int wid = blockIdx.x * 4 + (threadIdx.x >> 5);
    int lane = threadIdx.x & 31;
    int mt = wid >> 4, nq = wid & 15;
    v8f acc[4] = {};
    gemm_tile_16x64(A + (size_t)mt * 16 * CC, CC, W + (size_t)nq * 64 * CC, CC, CC, lane, acc);
    int nl = lane & 15, mh = lane >> 4;
#pragma unroll
    for (int j = 0; j < 4; ++j) {
        int n = nq * 64 + j * 16 + nl;
        float bv = bias[n];
#pragma unroll
        for (int v = 0; v < 8; ++v) {
            int m = mt * 16 + v + mh * 8;
            float x = acc[j][v] + bv;
            float ge = 0.5f * x * (1.f + erff(x * 0.70710678118654752f));
            g[(size_t)m * HID_ + n] = (_Float16)ge;
        }
    }
}

// ---------------- kernel 9: fc2 GEMM (N=256,K=1024) + residual -> d_out ----------------
__global__ void k_gemm_fc2(const _Float16* __restrict__ A, const _Float16* __restrict__ W,
                           const float* __restrict__ bias, const float* __restrict__ x2,
                           float* __restrict__ out) {
    int wid = blockIdx.x * 4 + (threadIdx.x >> 5);
    int lane = threadIdx.x & 31;
    int mt = wid >> 2, nq = wid & 3;
    v8f acc[4] = {};
    gemm_tile_16x64(A + (size_t)mt * 16 * HID_, HID_, W + (size_t)nq * 64 * HID_, HID_,
                    HID_, lane, acc);
    int nl = lane & 15, mh = lane >> 4;
#pragma unroll
    for (int j = 0; j < 4; ++j) {
        int n = nq * 64 + j * 16 + nl;
        float bv = bias[n];
#pragma unroll
        for (int v = 0; v < 8; ++v) {
            int m = mt * 16 + v + mh * 8;
            size_t idx = (size_t)m * CC + n;
            out[idx] = x2[idx] + acc[j][v] + bv;
        }
    }
}

// ---------------- host: workspace layout + launches ----------------
extern "C" void kernel_launch(void* const* d_in, const int* in_sizes, int n_in,
                              void* d_out, int out_size, void* d_ws, size_t ws_size,
                              hipStream_t stream) {
    (void)in_sizes; (void)n_in; (void)out_size; (void)ws_size;
    const float* x       = (const float*)d_in[0];
    const float* ape_w   = (const float*)d_in[1];
    const float* ape_b   = (const float*)d_in[2];
    const float* norm1_w = (const float*)d_in[3];
    const float* norm1_b = (const float*)d_in[4];
    const float* qkv_w   = (const float*)d_in[5];
    const float* qkv_b   = (const float*)d_in[6];
    const float* scale   = (const float*)d_in[7];
    const float* rpb1_w  = (const float*)d_in[8];
    const float* rpb1_b  = (const float*)d_in[9];
    const float* rpb2_w  = (const float*)d_in[10];
    const float* rpb2_b  = (const float*)d_in[11];
    const float* proj_w  = (const float*)d_in[12];
    const float* proj_b  = (const float*)d_in[13];
    const float* norm2_w = (const float*)d_in[14];
    const float* norm2_b = (const float*)d_in[15];
    const float* fc1_w   = (const float*)d_in[16];
    const float* fc1_b   = (const float*)d_in[17];
    const float* fc2_w   = (const float*)d_in[18];
    const float* fc2_b   = (const float*)d_in[19];
    float* outp = (float*)d_out;

    char* ws = (char*)d_ws;
    constexpr size_t NELEM = (size_t)MTOT * CC;                 // 12,845,056
    constexpr size_t XS_OFF   = 0;                              // f32 x+APE (shortcut)
    constexpr size_t X2_OFF   = XS_OFF + NELEM * 4;             // f32 after attn residual
    constexpr size_t WINA_OFF = X2_OFF + NELEM * 4;             // f16: winA -> attn_out -> h2
    constexpr size_t QKV_OFF  = WINA_OFF + NELEM * 2;           // f16: qkv -> later g (fc1 out)
    constexpr size_t WQ_OFF   = QKV_OFF + (size_t)MTOT * HID_ * 2; // g is the larger occupant
    constexpr size_t WP_OFF   = WQ_OFF + (size_t)768 * CC * 2;
    constexpr size_t W1_OFF   = WP_OFF + (size_t)CC * CC * 2;
    constexpr size_t W2_OFF   = W1_OFF + (size_t)HID_ * CC * 2;
    constexpr size_t TAB_OFF  = W2_OFF + (size_t)CC * HID_ * 2;

    float*     xs   = (float*)(ws + XS_OFF);
    float*     x2   = (float*)(ws + X2_OFF);
    _Float16*  winA = (_Float16*)(ws + WINA_OFF);   // also attn_out, then h2
    _Float16*  qkvb = (_Float16*)(ws + QKV_OFF);    // also g
    _Float16*  wq   = (_Float16*)(ws + WQ_OFF);
    _Float16*  wp   = (_Float16*)(ws + WP_OFF);
    _Float16*  w1   = (_Float16*)(ws + W1_OFF);
    _Float16*  w2   = (_Float16*)(ws + W2_OFF);
    float*     tab  = (float*)(ws + TAB_OFF);

    // weight conversions
    k_cvt<<<(768 * CC + 255) / 256, 256, 0, stream>>>(qkv_w, wq, 768 * CC);
    k_cvt<<<(CC * CC + 255) / 256, 256, 0, stream>>>(proj_w, wp, CC * CC);
    k_cvt<<<(HID_ * CC + 255) / 256, 256, 0, stream>>>(fc1_w, w1, HID_ * CC);
    k_cvt<<<(CC * HID_ + 255) / 256, 256, 0, stream>>>(fc2_w, w2, CC * HID_);

    // 1. APE conv + residual
    k_conv<<<(int)(NELEM / 256), 256, 0, stream>>>(x, ape_w, ape_b, xs);
    // 2. LN1 + shift + window gather -> f16
    k_ln1win<<<MTOT, 256, 0, stream>>>(xs, norm1_w, norm1_b, winA);
    // 3. QKV GEMM (3136 M-tiles x 12 N-quads, 4 waves/block)
    k_gemm_qkv<<<(3136 * 12) / 4, 128, 0, stream>>>(winA, wq, qkv_b, qkvb);
    // 4. relative position bias table
    k_rpb<<<(169 * NHD + 255) / 256, 256, 0, stream>>>(rpb1_w, rpb1_b, rpb2_w, rpb2_b, tab);
    // 5. attention; attn_out aliases winA (winA fully consumed by k_gemm_qkv)
    k_attn<<<TOTW * NHD, 128, 0, stream>>>(qkvb, scale, tab, winA);
    // 6. proj GEMM + window reverse + roll + residual
    k_gemm_proj<<<(3136 * 4) / 4, 128, 0, stream>>>(winA, wp, proj_b, xs, x2);
    // 7. LN2 -> f16 (h2 aliases winA; attn_out dead after proj)
    k_ln2<<<MTOT, 256, 0, stream>>>(x2, norm2_w, norm2_b, winA);
    // 8. fc1 + GELU (g aliases qkv region; qkv dead after attention)
    k_gemm_fc1<<<(3136 * 16) / 4, 128, 0, stream>>>(winA, w1, fc1_b, qkvb);
    // 9. fc2 + residual -> output
    k_gemm_fc2<<<(3136 * 4) / 4, 128, 0, stream>>>(qkvb, w2, fc2_b, x2, outp);
}
